// TemporalLogicLayer_76862734729472
// MI455X (gfx1250) — compile-verified, matrix operand order checked by reading, not run on hardware
//
#include <hip/hip_runtime.h>
#include <hip/hip_bf16.h>

typedef __attribute__((ext_vector_type(16))) _Float16 v16h;
typedef __attribute__((ext_vector_type(8)))  float    v8f;

// Problem constants
#define BB   32
#define TT   128
#define DD   128      // feature half-width; X has 2*DD = 256 features
#define HP   160      // H=132 padded to 160 = 5 k-tiles of 32 = 10 n-tiles of 16
#define OO   64

#define KT1  8        // 256/32 k-tiles for GEMM1
#define NT1  10       // 160/16 n-tiles
#define KT2  5        // 160/32
#define NT2  10
#define KT3  5
#define NT3  4        // 64/16

// Swizzled f16 weight sizes (halves): [kt][nt][lane(32)][16]
#define W1S_N (KT1*NT1*32*16)   // 40960
#define W2S_N (KT2*NT2*32*16)   // 25600
#define W3S_N (KT3*NT3*32*16)   // 10240

// ---- fragment swizzle for v_wmma_f32_16x16x32_f16 (16-bit operand layout) ----
// ISA 16-bit A 16x32 table: lane = m + 16*g ; half index h = 2*v+p ;
// k = (v<4 ? 2v : 2v+8) + 8g + p
__device__ __forceinline__ int kk_from_hvg(int v, int p, int g) {
    return (v < 4 ? 2 * v : 2 * v + 8) + 8 * g + p;
}
// inverse: kk (0..31) -> (g, v, p)
__device__ __forceinline__ int swzA(int r, int k, int KT) {
    int rt = r >> 4, m = r & 15;
    int kt = k >> 5, kk = k & 31;
    int hi = kk >> 4, lo = kk & 15;
    int g = lo >> 3;
    int v = hi * 4 + ((lo & 7) >> 1);
    int p = kk & 1;
    return (((rt * KT + kt) * 32) + (m + 16 * g)) * 16 + 2 * v + p;
}

// ---------------- weight prep: fp32 -> f16, zero-padded, fragment-swizzled ----
__global__ void tll_prep_kernel(const float* __restrict__ W1, const float* __restrict__ b1,
                                const float* __restrict__ W2, const float* __restrict__ b2,
                                const float* __restrict__ W3, const float* __restrict__ b3,
                                _Float16* __restrict__ W1s, _Float16* __restrict__ W2s,
                                _Float16* __restrict__ W3s,
                                float* __restrict__ b1p, float* __restrict__ b2p,
                                float* __restrict__ b3p) {
    int idx = blockIdx.x * blockDim.x + threadIdx.x;

    const float* src;
    _Float16* dst;
    int e, NT, Krows, Ncols, ld;
    if (idx < W1S_N) {
        e = idx;              src = W1; dst = W1s; NT = NT1; Krows = 2 * DD; Ncols = 132; ld = 132;
    } else if (idx < W1S_N + W2S_N) {
        e = idx - W1S_N;      src = W2; dst = W2s; NT = NT2; Krows = 132;    Ncols = 132; ld = 132;
    } else if (idx < W1S_N + W2S_N + W3S_N) {
        e = idx - W1S_N - W2S_N; src = W3; dst = W3s; NT = NT3; Krows = 132; Ncols = 64; ld = 64;
    } else {
        int i2 = idx - (W1S_N + W2S_N + W3S_N);
        if (i2 < HP)            b1p[i2]        = (i2 < 132) ? b1[i2] : 0.0f;
        else if (i2 < 2 * HP)   b2p[i2 - HP]   = (i2 - HP < 132) ? b2[i2 - HP] : 0.0f;
        else if (i2 < 2 * HP + OO) b3p[i2 - 2 * HP] = b3[i2 - 2 * HP];
        return;
    }
    int h = e & 15, lane = (e >> 4) & 31, frag = e >> 9;
    int kt = frag / NT, nt = frag % NT;
    int g = lane >> 4, nl = lane & 15, v = h >> 1, p = h & 1;
    int k = kt * 32 + kk_from_hvg(v, p, g);
    int n = nt * 16 + nl;
    float val = (k < Krows && n < Ncols) ? src[k * ld + n] : 0.0f;
    dst[e] = (_Float16)val;
}

// ---- software-pipelined tile GEMM: prefetch depth 1 on A and B fragments ----
template<int KT, int NT>
__device__ __forceinline__ void gemm_tile(const _Float16* sA,              // LDS, swizzled A frags
                                          const _Float16* __restrict__ Wt, // global, swizzled B frags
                                          int w, int lane, v8f* acc) {
    const _Float16* aBase = &sA[((w * KT) * 32 + lane) * 16];
    const _Float16* bBase = &Wt[lane * 16];

    v16h a  = *(const v16h*)&aBase[0];
    v16h b  = *(const v16h*)&bBase[0];
    v16h an = a, bn = b;
    #pragma unroll
    for (int i = 0; i < KT * NT; ++i) {
        const int nt = i % NT;
        if (i + 1 < KT * NT) {
            // issue next-iteration loads BEFORE the WMMA consuming this one
            bn = *(const v16h*)&bBase[(size_t)(i + 1) * 32 * 16];
            if ((i + 1) % NT == 0)
                an = *(const v16h*)&aBase[(size_t)((i + 1) / NT) * 32 * 16];
        }
        acc[nt] = __builtin_amdgcn_wmma_f32_16x16x32_f16(
            false, a, false, b, (short)0, acc[nt], false, false);
        a = an; b = bn;
    }
}

// ---------------- main kernel: one workgroup per (t, batch) -------------------
__global__ __launch_bounds__(256, 2)
void tll_main_kernel(const float* __restrict__ P,
                     const _Float16* __restrict__ W1s, const _Float16* __restrict__ W2s,
                     const _Float16* __restrict__ W3s,
                     const float* __restrict__ b1p, const float* __restrict__ b2p,
                     const float* __restrict__ b3p,
                     float* __restrict__ out) {
    __shared__ _Float16 sX[8 * KT1 * 32 * 16];   // 32768 halves = 64 KB (X, swizzled A frags)
    __shared__ _Float16 sH[8 * KT2 * 32 * 16];   // 20480 halves = 40 KB (H1/H2, reused)
    __shared__ float    sPart[16 * OO];          // [wave*2+g][col] partial maxima, 4 KB

    const int t     = blockIdx.x;
    const int batch = blockIdx.y;
    const int tid   = threadIdx.x;
    const int R     = TT - t;                    // valid rows s = t..T-1
    const float* Pb = P + (size_t)batch * TT * DD;

    // init reduction buffer
    for (int i = tid; i < 16 * OO; i += 256) sPart[i] = -1.0e30f;

    // ---- Phase A: build swizzled f16 X in LDS -------------------------------
    if (tid < DD) {
        const int d = tid;                       // cummax channels -> k = d
        float rm = -3.0e38f;
        for (int r = 0; r < R; ++r) {
            float pv = Pb[(t + r) * DD + d];
            rm = fmaxf(rm, pv);
            sX[swzA(r, d, KT1)] = (_Float16)rm;
        }
        for (int r = R; r < TT; ++r) sX[swzA(r, d, KT1)] = (_Float16)0.0f;
    } else {
        const int d = tid - DD;                  // endpoint channels -> k = 128 + d
        for (int r = 0; r < R; ++r)
            sX[swzA(r, DD + d, KT1)] = (_Float16)Pb[(t + r) * DD + d];
        for (int r = R; r < TT; ++r) sX[swzA(r, DD + d, KT1)] = (_Float16)0.0f;
    }
    __syncthreads();

    const int w = tid >> 5, lane = tid & 31;
    const int g = lane >> 4, nl = lane & 15;

    if (w * 16 < R) {                            // row-tile has at least one valid row
        // ---- GEMM1: X(128x256) @ W1s(256x160) -------------------------------
        v8f acc1[NT1];
        #pragma unroll
        for (int nt = 0; nt < NT1; ++nt) acc1[nt] = (v8f){0,0,0,0,0,0,0,0};
        gemm_tile<KT1, NT1>(sX, W1s, w, lane, acc1);

        // bias + relu -> swizzled H1 (own rows only: no barrier needed)
        #pragma unroll
        for (int nt = 0; nt < NT1; ++nt) {
            int n = nt * 16 + nl;
            float bb = b1p[n];
            #pragma unroll
            for (int j = 0; j < 8; ++j) {
                float hv = fmaxf(acc1[nt][j] + bb, 0.0f);
                int r = w * 16 + j + 8 * g;
                sH[swzA(r, n, KT2)] = (_Float16)hv;
            }
        }

        // ---- GEMM2: H1(128x160) @ W2s(160x160) ------------------------------
        v8f acc2[NT2];
        #pragma unroll
        for (int nt = 0; nt < NT2; ++nt) acc2[nt] = (v8f){0,0,0,0,0,0,0,0};
        gemm_tile<KT2, NT2>(sH, W2s, w, lane, acc2);

        #pragma unroll
        for (int nt = 0; nt < NT2; ++nt) {
            int n = nt * 16 + nl;
            float bb = b2p[n];
            #pragma unroll
            for (int j = 0; j < 8; ++j) {
                float hv = fmaxf(acc2[nt][j] + bb, 0.0f);
                int r = w * 16 + j + 8 * g;
                sH[swzA(r, n, KT2)] = (_Float16)hv;    // overwrite own rows with H2
            }
        }

        // ---- GEMM3: H2(128x160) @ W3s(160x64) -------------------------------
        v8f acc3[NT3];
        #pragma unroll
        for (int nt = 0; nt < NT3; ++nt) acc3[nt] = (v8f){0,0,0,0,0,0,0,0};
        gemm_tile<KT3, NT3>(sH, W3s, w, lane, acc3);

        // sigmoid(5x), mask invalid rows, per-lane column max
        #pragma unroll
        for (int nt = 0; nt < NT3; ++nt) {
            int c = nt * 16 + nl;
            float bb = b3p[c];
            float cm = -1.0e30f;
            #pragma unroll
            for (int j = 0; j < 8; ++j) {
                int r = w * 16 + j + 8 * g;
                float x = (acc3[nt][j] + bb) * 5.0f;
                float y = 1.0f / (1.0f + __expf(-x));
                cm = fmaxf(cm, (r < R) ? y : -1.0e30f);
            }
            sPart[(w * 2 + g) * OO + c] = cm;
        }
    }
    __syncthreads();

    // ---- final cross-wave max over 16 partials per output column ------------
    if (tid < OO) {
        float m = -1.0e30f;
        #pragma unroll
        for (int i = 0; i < 16; ++i) m = fmaxf(m, sPart[i * OO + tid]);
        out[((size_t)batch * TT + t) * OO + tid] = m;
    }
}

extern "C" void kernel_launch(void* const* d_in, const int* in_sizes, int n_in,
                              void* d_out, int out_size, void* d_ws, size_t ws_size,
                              hipStream_t stream) {
    const float* P  = (const float*)d_in[0];
    const float* W1 = (const float*)d_in[1];
    const float* b1 = (const float*)d_in[2];
    const float* W2 = (const float*)d_in[3];
    const float* b2 = (const float*)d_in[4];
    const float* W3 = (const float*)d_in[5];
    const float* b3 = (const float*)d_in[6];

    _Float16* W1s = (_Float16*)d_ws;
    _Float16* W2s = W1s + W1S_N;
    _Float16* W3s = W2s + W2S_N;
    float*    b1p = (float*)(W3s + W3S_N);      // byte offset 153600, 4B aligned
    float*    b2p = b1p + HP;
    float*    b3p = b2p + HP;

    const int prepN = W1S_N + W2S_N + W3S_N + 2 * HP + OO;
    tll_prep_kernel<<<(prepN + 255) / 256, 256, 0, stream>>>(
        W1, b1, W2, b2, W3, b3, W1s, W2s, W3s, b1p, b2p, b3p);

    tll_main_kernel<<<dim3(TT, BB), 256, 0, stream>>>(
        P, W1s, W2s, W3s, b1p, b2p, b3p, (float*)d_out);
}
